// EEGMamba_78958678769770
// MI455X (gfx1250) — compile-verified, matrix operand order checked by reading, not run on hardware
//
#include <hip/hip_runtime.h>
#include <math.h>

// ---- model dims ----
#define Bsz  8
#define Cch  16
#define Tlen 1024
#define DMd  128
#define DId  256
#define DSs  16
#define DRr  8
#define DCc  4
#define Lnum 4
#define NXP  (DRr + 2*DSs)   // 40

typedef __bf16 bf16_t;
typedef __attribute__((ext_vector_type(16))) __bf16 v16bf;
typedef __attribute__((ext_vector_type(8)))  float  v8f;

__device__ __forceinline__ bf16_t f2bf(float f) {
    unsigned u = __builtin_bit_cast(unsigned, f);
    unsigned r = (u + 0x7FFFu + ((u >> 16) & 1u)) >> 16;   // round-to-nearest-even
    unsigned short s = (unsigned short)r;
    return __builtin_bit_cast(bf16_t, s);
}

// K offset inside a 32-wide k-tile for WMMA bf16 A/B fragments (per ISA layout).
__device__ __forceinline__ int frag_kk(int half, int j) {
    int v = j >> 1, lo = j & 1;
    return (v < 4 ? 2 * v : 16 + 2 * (v - 4)) + 8 * half + lo;
}

// ---------------- weight shuffle: W[K,N] f32 -> fragment-native bf16 ----------------
// out[ktile*(N*32) + n*32 + half*16 + j] = bf16( W[(ktile*32 + kk(half,j)) * N + n] )
// => in the GEMM each lane reads 16 CONTIGUOUS bf16 (32B) per B fragment.
__global__ void k_shuffle_w(const float* __restrict__ W, bf16_t* __restrict__ out, int N, int K) {
    int idx = blockIdx.x * blockDim.x + threadIdx.x;
    if (idx >= K * N) return;
    int ktile = idx / (N * 32);
    int rem   = idx % (N * 32);
    int n     = rem / 32;
    int q     = rem % 32;
    int half = q >> 4, j = q & 15;
    int kk = frag_kk(half, j);
    out[idx] = f2bf(W[(size_t)(ktile * 32 + kk) * N + n]);
}

// ---------------- input projection: (B,T,C) @ (C,DM) + b ----------------
__global__ void k_input_proj(const float* __restrict__ eeg, const float* __restrict__ Win,
                             const float* __restrict__ b_in, float* __restrict__ h) {
    int row = blockIdx.x;            // b*T + t
    int m   = threadIdx.x;           // 0..DM-1
    int b = row / Tlen, t = row % Tlen;
    float acc = b_in[m];
#pragma unroll
    for (int c = 0; c < Cch; c++)
        acc += eeg[((size_t)b * Cch + c) * Tlen + t] * Win[c * DMd + m];
    h[(size_t)row * DMd + m] = acc;
}

// ---------------- LayerNorm over DM, emit bf16 ----------------
__global__ void k_ln_bf16(const float* __restrict__ h, const float* __restrict__ w,
                          const float* __restrict__ bb, bf16_t* __restrict__ out) {
    __shared__ float red[DMd];
    int row = blockIdx.x, m = threadIdx.x;
    float x = h[(size_t)row * DMd + m];
    red[m] = x; __syncthreads();
    for (int s = DMd / 2; s > 0; s >>= 1) { if (m < s) red[m] += red[m + s]; __syncthreads(); }
    float mean = red[0] * (1.f / DMd); __syncthreads();
    float d = x - mean;
    red[m] = d * d; __syncthreads();
    for (int s = DMd / 2; s > 0; s >>= 1) { if (m < s) red[m] += red[m + s]; __syncthreads(); }
    float var = red[0] * (1.f / DMd);
    out[(size_t)row * DMd + m] = f2bf(d * rsqrtf(var + 1e-5f) * w[m] + bb[m]);
}

// ---------------- WMMA bf16 GEMM: C[M,N] (+)= A[M,K] @ W[K,N] (+ bias) ----------------
// one wave per 32x32 output tile (2x2 grid of 16x16 WMMA tiles).
// A: row-major bf16 [M,K]; Wsh: fragment-shuffled bf16 (see k_shuffle_w).
// K compile-time -> fully unrolled. __launch_bounds__(256,1) relaxes the VGPR
// budget so the pipelined fragment loads don't spill (grids here have 1k-4k
// waves, so reduced per-SIMD occupancy still saturates the device).
template<int K, bool ADD, bool BIAS>
__global__ __launch_bounds__(256, 1)
void k_wmma_gemm(const bf16_t* __restrict__ A, const bf16_t* __restrict__ Wsh,
                 const float* __restrict__ bias, float* __restrict__ C,
                 int M, int N) {
    int gwave = (blockIdx.x * blockDim.x + threadIdx.x) >> 5;
    int lane  = threadIdx.x & 31;
    int ntiles = N >> 5;                 // 32-wide tiles
    int mt = gwave / ntiles;
    int nt = gwave - mt * ntiles;
    if (mt >= (M >> 5)) return;          // wave-uniform exit: live waves keep EXEC all-1

    int half = lane >> 4;
    int r    = lane & 15;

    v8f acc00, acc01, acc10, acc11;
#pragma unroll
    for (int v = 0; v < 8; v++) { acc00[v] = 0.f; acc01[v] = 0.f; acc10[v] = 0.f; acc11[v] = 0.f; }

    const bf16_t* Arow0 = A + (size_t)(mt * 32 + r) * K;
    const bf16_t* Arow1 = Arow0 + (size_t)16 * K;
    const bf16_t* Wl0 = Wsh + (size_t)(nt * 32 + r) * 32 + half * 16;          // n-subtile 0
    const bf16_t* Wl1 = Wl0 + (size_t)16 * 32;                                  // n-subtile 1

#pragma unroll
    for (int k0 = 0; k0 < K; k0 += 32) {
        size_t ktoff = (size_t)(k0 >> 5) * ((size_t)N * 32);
        v16bf a0, a1, b0, b1;
#pragma unroll
        for (int j = 0; j < 16; j++) {
            int kk = frag_kk(half, j) + k0;
            a0[j] = Arow0[kk];
            a1[j] = Arow1[kk];
            b0[j] = Wl0[ktoff + j];      // contiguous 32B per fragment
            b1[j] = Wl1[ktoff + j];
        }
        acc00 = __builtin_amdgcn_wmma_f32_16x16x32_bf16(false, a0, false, b0, (short)0, acc00, false, false);
        acc01 = __builtin_amdgcn_wmma_f32_16x16x32_bf16(false, a0, false, b1, (short)0, acc01, false, false);
        acc10 = __builtin_amdgcn_wmma_f32_16x16x32_bf16(false, a1, false, b0, (short)0, acc10, false, false);
        acc11 = __builtin_amdgcn_wmma_f32_16x16x32_bf16(false, a1, false, b1, (short)0, acc11, false, false);
    }

    int n0 = nt * 32 + r, n1 = n0 + 16;
    float bv0 = BIAS ? bias[n0] : 0.f;
    float bv1 = BIAS ? bias[n1] : 0.f;
#pragma unroll
    for (int v = 0; v < 8; v++) {
        int m0 = mt * 32 + v + 8 * half;
        int m1 = m0 + 16;
        size_t i00 = (size_t)m0 * N + n0, i01 = (size_t)m0 * N + n1;
        size_t i10 = (size_t)m1 * N + n0, i11 = (size_t)m1 * N + n1;
        if (ADD) {
            C[i00] += acc00[v] + bv0;  C[i01] += acc01[v] + bv1;
            C[i10] += acc10[v] + bv0;  C[i11] += acc11[v] + bv1;
        } else {
            C[i00] = acc00[v] + bv0;   C[i01] = acc01[v] + bv1;
            C[i10] = acc10[v] + bv0;   C[i11] = acc11[v] + bv1;
        }
    }
}

// ---------------- depthwise causal conv (+bias +SiLU); rev flips causality ----------------
__global__ void k_conv_silu(const float* __restrict__ xz, const float* __restrict__ cw,
                            const float* __restrict__ cb, float* __restrict__ xc, int rev) {
    int idx = blockIdx.x * blockDim.x + threadIdx.x;
    if (idx >= Bsz * Tlen * DId) return;
    int d = idx % DId;
    int t = (idx / DId) % Tlen;
    int b = idx / (DId * Tlen);
    float acc = cb[d];
#pragma unroll
    for (int j = 0; j < DCc; j++) {
        int tt = rev ? (t + (DCc - 1) - j) : (t + j - (DCc - 1));
        if (tt >= 0 && tt < Tlen)
            acc += cw[d * DCc + j] * xz[((size_t)b * Tlen + tt) * (2 * DId) + d];
    }
    xc[idx] = acc / (1.f + __expf(-acc));       // SiLU
}

// ---------------- small-N f32 GEMM (one row per block); act=1 -> softplus ----------------
__global__ void k_small_gemm(const float* __restrict__ A, int lda,
                             const float* __restrict__ W, const float* __restrict__ bias,
                             float* __restrict__ C, int N, int K, int act) {
    int row = blockIdx.x;
    int n = threadIdx.x;
    if (n >= N) return;
    float acc = bias ? bias[n] : 0.f;
    const float* a = A + (size_t)row * lda;
    for (int k = 0; k < K; k++) acc += a[k] * W[k * N + n];
    if (act == 1) acc = (acc > 20.f) ? acc : log1pf(__expf(acc));   // softplus
    C[(size_t)row * N + n] = acc;
}

// ---------------- selective scan: 16 lanes = 16 SSM states of one channel ----------------
__global__ void k_scan(const float* __restrict__ delta, const float* __restrict__ xc,
                       const float* __restrict__ xdbl, const float* __restrict__ A_log,
                       float* __restrict__ y, int rev) {
    int s    = threadIdx.x & 15;
    int dloc = threadIdx.x >> 4;                // 0..15 channels per block
    int dblk = blockIdx.x % (DId / 16);
    int b    = blockIdx.x / (DId / 16);
    int d    = dblk * 16 + dloc;
    float a = -__expf(A_log[d * DSs + s]);
    float h = 0.f;
    for (int i = 0; i < Tlen; i++) {
        int t = rev ? (Tlen - 1 - i) : i;
        size_t row = (size_t)b * Tlen + t;
        float dl = delta[row * DId + d];
        float xv = xc[row * DId + d];
        float Bs = xdbl[row * NXP + DRr + s];
        float Cs = xdbl[row * NXP + DRr + DSs + s];
        h = __expf(dl * a) * h + dl * Bs * xv;
        float p = h * Cs;
        p += __shfl_xor(p, 1, 16);
        p += __shfl_xor(p, 2, 16);
        p += __shfl_xor(p, 4, 16);
        p += __shfl_xor(p, 8, 16);
        if (s == 0) y[row * DId + d] = p;
    }
}

// ---------------- gate: (y + Dp*x) * silu(z) -> bf16 ----------------
__global__ void k_gate(const float* __restrict__ y, const float* __restrict__ xc,
                       const float* __restrict__ xz, const float* __restrict__ Dp,
                       bf16_t* __restrict__ out) {
    int idx = blockIdx.x * blockDim.x + threadIdx.x;
    if (idx >= Bsz * Tlen * DId) return;
    int d = idx % DId;
    size_t row = (size_t)idx / DId;
    float z = xz[row * (2 * DId) + DId + d];
    float g = z / (1.f + __expf(-z));
    float v = y[idx] + Dp[d] * xc[idx];
    out[idx] = f2bf(v * g);
}

// ---------------- head: mean-pool, LN, MLP(128->64 gelu ->1) ----------------
__global__ void k_head(const float* __restrict__ h, const float* __restrict__ lnw,
                       const float* __restrict__ lnb, const float* __restrict__ W1,
                       const float* __restrict__ b1, const float* __restrict__ W2,
                       const float* __restrict__ b2, float* __restrict__ out) {
    __shared__ float red[DMd];
    __shared__ float p[DMd];
    __shared__ float q[DMd / 2];
    int b = blockIdx.x, m = threadIdx.x;
    float s = 0.f;
    for (int t = 0; t < Tlen; t++) s += h[((size_t)b * Tlen + t) * DMd + m];
    s *= (1.f / Tlen);
    red[m] = s; __syncthreads();
    for (int k = DMd / 2; k > 0; k >>= 1) { if (m < k) red[m] += red[m + k]; __syncthreads(); }
    float mean = red[0] * (1.f / DMd); __syncthreads();
    float dd = s - mean;
    red[m] = dd * dd; __syncthreads();
    for (int k = DMd / 2; k > 0; k >>= 1) { if (m < k) red[m] += red[m + k]; __syncthreads(); }
    float var = red[0] * (1.f / DMd);
    p[m] = dd * rsqrtf(var + 1e-5f) * lnw[m] + lnb[m];
    __syncthreads();
    if (m < DMd / 2) {
        float acc = b1[m];
        for (int k = 0; k < DMd; k++) acc += p[k] * W1[k * (DMd / 2) + m];
        float x3 = acc * acc * acc;          // tanh-approx GELU (jax default)
        q[m] = 0.5f * acc * (1.f + tanhf(0.7978845608028654f * (acc + 0.044715f * x3)));
    }
    __syncthreads();
    if (m == 0) {
        float acc = b2[0];
        for (int k = 0; k < DMd / 2; k++) acc += q[k] * W2[k];
        out[b] = acc;
    }
}

extern "C" void kernel_launch(void* const* d_in, const int* in_sizes, int n_in,
                              void* d_out, int out_size, void* d_ws, size_t ws_size,
                              hipStream_t stream) {
    (void)in_sizes; (void)n_in; (void)out_size; (void)ws_size;
    const float* eeg    = (const float*)d_in[0];
    const float* Win    = (const float*)d_in[1];
    const float* b_in   = (const float*)d_in[2];
    const float* ln_w   = (const float*)d_in[3];
    const float* ln_b   = (const float*)d_in[4];
    const float* in_w   = (const float*)d_in[5];
    const float* conv_w = (const float*)d_in[6];
    const float* conv_b = (const float*)d_in[7];
    const float* xp_w   = (const float*)d_in[8];
    const float* dt_w   = (const float*)d_in[9];
    const float* dt_b   = (const float*)d_in[10];
    const float* A_log  = (const float*)d_in[11];
    const float* Dp     = (const float*)d_in[12];
    const float* out_w  = (const float*)d_in[13];
    const float* out_b  = (const float*)d_in[14];
    const float* clnw   = (const float*)d_in[15];
    const float* clnb   = (const float*)d_in[16];
    const float* W1     = (const float*)d_in[17];
    const float* b1     = (const float*)d_in[18];
    const float* W2     = (const float*)d_in[19];
    const float* b2     = (const float*)d_in[20];

    const int ROWS = Bsz * Tlen;   // 8192
    char* ws = (char*)d_ws;
    size_t off = 0;
    auto alloc = [&](size_t bytes) -> char* {
        char* p = ws + off; off += (bytes + 255) & ~(size_t)255; return p;
    };
    bf16_t* in_w_bf  = (bf16_t*)alloc((size_t)Lnum * 2 * DMd * 2 * DId * 2);  // shuffled
    bf16_t* out_w_bf = (bf16_t*)alloc((size_t)Lnum * 2 * DId * DMd * 2);      // shuffled
    float*  hbuf     = (float*) alloc((size_t)ROWS * DMd * 4);
    bf16_t* xn_bf    = (bf16_t*)alloc((size_t)ROWS * DMd * 2);
    float*  xzbuf    = (float*) alloc((size_t)ROWS * 2 * DId * 4);
    float*  xconv    = (float*) alloc((size_t)ROWS * DId * 4);
    float*  xdbl     = (float*) alloc((size_t)ROWS * NXP * 4);
    float*  delta    = (float*) alloc((size_t)ROWS * DId * 4);
    float*  yscan    = (float*) alloc((size_t)ROWS * DId * 4);
    bf16_t* ygate    = (bf16_t*)alloc((size_t)ROWS * DId * 2);

    // shuffle+convert all weight matrices into WMMA fragment layout
    for (int ld = 0; ld < Lnum * 2; ld++) {
        { int n = DMd * 2 * DId;   // in_w[ld]: K=128, N=512
          k_shuffle_w<<<(n + 255) / 256, 256, 0, stream>>>(
              in_w + (size_t)ld * DMd * 2 * DId, in_w_bf + (size_t)ld * DMd * 2 * DId, 2 * DId, DMd); }
        { int n = DId * DMd;       // out_w[ld]: K=256, N=128
          k_shuffle_w<<<(n + 255) / 256, 256, 0, stream>>>(
              out_w + (size_t)ld * DId * DMd, out_w_bf + (size_t)ld * DId * DMd, DMd, DId); }
    }

    k_input_proj<<<ROWS, DMd, 0, stream>>>(eeg, Win, b_in, hbuf);

    for (int l = 0; l < Lnum; l++) {
        k_ln_bf16<<<ROWS, DMd, 0, stream>>>(hbuf, ln_w + l * DMd, ln_b + l * DMd, xn_bf);
        for (int dir = 0; dir < 2; dir++) {
            int ld = l * 2 + dir;
            {   // xz = xn @ in_w   (M=8192, N=512, K=128)
                int M = ROWS, N = 2 * DId;
                int waves = (M / 32) * (N / 32);
                k_wmma_gemm<DMd, false, false><<<(waves + 7) / 8, 256, 0, stream>>>(
                    xn_bf, in_w_bf + (size_t)ld * DMd * 2 * DId, nullptr, xzbuf, M, N);
            }
            {   int n = ROWS * DId;
                k_conv_silu<<<(n + 255) / 256, 256, 0, stream>>>(
                    xzbuf, conv_w + (size_t)ld * DId * DCc, conv_b + (size_t)ld * DId, xconv, dir);
            }
            // xdbl = x @ xp_w  (N=40, K=256)
            k_small_gemm<<<ROWS, 64, 0, stream>>>(
                xconv, DId, xp_w + (size_t)ld * DId * NXP, nullptr, xdbl, NXP, DId, 0);
            // delta = softplus(dt @ dt_w + dt_b)  (N=256, K=8)
            k_small_gemm<<<ROWS, DId, 0, stream>>>(
                xdbl, NXP, dt_w + (size_t)ld * DRr * DId, dt_b + (size_t)ld * DId, delta, DId, DRr, 1);
            k_scan<<<Bsz * (DId / 16), 256, 0, stream>>>(
                delta, xconv, xdbl, A_log + (size_t)ld * DId * DSs, yscan, dir);
            {   int n = ROWS * DId;
                k_gate<<<(n + 255) / 256, 256, 0, stream>>>(
                    yscan, xconv, xzbuf, Dp + (size_t)ld * DId, ygate);
            }
            {   // hbuf += ygate @ out_w + out_b  (M=8192, N=128, K=256)
                int M = ROWS, N = DMd;
                int waves = (M / 32) * (N / 32);
                k_wmma_gemm<DId, true, true><<<(waves + 7) / 8, 256, 0, stream>>>(
                    ygate, out_w_bf + (size_t)ld * DId * DMd, out_b + (size_t)ld * DMd, hbuf, M, N);
            }
        }
    }
    k_head<<<Bsz, DMd, 0, stream>>>(hbuf, clnw, clnb, W1, b1, W2, b2, (float*)d_out);
}
